// NeuralODE_35708358099537
// MI455X (gfx1250) — compile-verified, matrix operand order checked by reading, not run on hardware
//
#include <hip/hip_runtime.h>
#include <math.h>
#include <stdint.h>

// Problem constants (match reference)
#define B_TOT   2048
#define L_DIM   128
#define P_DIM   8
#define W_DIM   512
#define T_DIM   128
#define K0PAD   160     // 128+8 = 136 padded up to 5 k-tiles of 32
#define M_TILE  16      // batch rows per workgroup
#define NWAVES  8
#define NTHREADS 256

typedef __attribute__((ext_vector_type(16))) __bf16 v16bf;
typedef __attribute__((ext_vector_type(8)))  __bf16 bf16x8;
typedef __attribute__((ext_vector_type(8)))  float  v8f;

// Branch-free tanh on the hardware transcendental pipe:
// tanh(x) = 1 - 2/(exp(2x)+1);  exp(2x) = exp2(x * 2*log2(e)).
// Saturates to +/-1 for large |x| (exp2 -> inf/0, rcp(inf)=0), NaN propagates.
__device__ __forceinline__ float fast_tanh(float x) {
  float e = __builtin_amdgcn_exp2f(x * 2.8853900817779268f);
  return 1.f - 2.f * __builtin_amdgcn_rcpf(e + 1.f);
}

// Load a 16x32 bf16 WMMA fragment (A or B operand) from a row-major
// [rows][ld] bf16 array. Per CDNA5 ISA 16-bit A layout:
//   lane l -> row (l & 15); K chunks [c0, c0+8) and [c0+16, c0+24),
//   c0 = (l < 16 ? 0 : 8), relative to kt*32.
__device__ __forceinline__ v16bf load_frag(const __bf16* base, int ld,
                                           int row0, int kt, int lane) {
  int r  = row0 + (lane & 15);
  int k0 = (kt << 5) + ((lane >> 4) << 3);
  const __bf16* p = base + r * ld + k0;
  bf16x8 lo = *(const bf16x8*)(p);
  bf16x8 hi = *(const bf16x8*)(p + 16);
  v16bf a;
#pragma unroll
  for (int i = 0; i < 8; ++i) { a[i] = lo[i]; a[i + 8] = hi[i]; }
  return a;
}

__device__ __forceinline__ v8f wmma_bf16(v16bf a, v16bf b, v8f c) {
  return __builtin_amdgcn_wmma_f32_16x16x32_bf16(
      false, a, false, b, (short)0, c, false, false);
}

// One dense layer: dst[M_TILE][N] = act(src[M_TILE][K] @ W[N][K]^T + bias)
// src is LDS bf16 (row stride ldsrc), W is global bf16 (row-major [N][K]),
// bias f32. K/N compile-time -> straight-line control flow; unrolling is
// deliberately limited (unroll 1 over job pairs, unroll 2 over K) to keep
// VGPR pressure low so no spilling and full multi-wave occupancy.
template <int K, int N, bool TANH, bool DST_BF16>
__device__ __forceinline__ void mlp_layer(const __bf16* src, int ldsrc,
                                          const __bf16* Wg, const float* bias,
                                          void* dstv, int lddst,
                                          int wave, int lane) {
  constexpr int ntiles = N >> 4;
  constexpr int kts    = K >> 5;
  constexpr int jobs   = ntiles / NWAVES;   // 4 (N=512) or 1 (N=128)
  constexpr int pairs  = jobs / 2;          // dual-accumulator passes

  auto epi = [&](v8f c, int nt2) {
    // C layout: vgpr r, lanes 0-15 -> (M=r, N=lane); lanes 16-31 -> (M=8+r, N=lane-16)
    const int col = (nt2 << 4) + (lane & 15);
    const int rb  = (lane >> 4) << 3;
    const float bi = bias[col];
#pragma unroll
    for (int r = 0; r < 8; ++r) {
      float v = c[r] + bi;
      if (TANH) v = fast_tanh(v);
      if (DST_BF16) ((__bf16*)dstv)[(rb + r) * lddst + col] = (__bf16)v;
      else          ((float*) dstv)[(rb + r) * lddst + col] = v;
    }
  };

#pragma unroll 1
  for (int jp = 0; jp < pairs; ++jp) {
    const int nt  = wave + (2 * jp) * NWAVES;
    const int ntB = nt + NWAVES;
    v8f c0 = {0.f, 0.f, 0.f, 0.f, 0.f, 0.f, 0.f, 0.f};
    v8f c1 = {0.f, 0.f, 0.f, 0.f, 0.f, 0.f, 0.f, 0.f};
#pragma unroll 2
    for (int kt = 0; kt < kts; ++kt) {
      v16bf a  = load_frag(src, ldsrc, 0, kt, lane);
      v16bf b0 = load_frag(Wg, K, nt << 4, kt, lane);
      c0 = wmma_bf16(a, b0, c0);
      v16bf b1 = load_frag(Wg, K, ntB << 4, kt, lane);
      c1 = wmma_bf16(a, b1, c1);
    }
    epi(c0, nt);
    epi(c1, ntB);
  }
  if constexpr (jobs & 1) {
    const int nt = wave + (jobs - 1) * NWAVES;
    v8f c0 = {0.f, 0.f, 0.f, 0.f, 0.f, 0.f, 0.f, 0.f};
#pragma unroll 2
    for (int kt = 0; kt < kts; ++kt) {
      v16bf a  = load_frag(src, ldsrc, 0, kt, lane);
      v16bf b0 = load_frag(Wg, K, nt << 4, kt, lane);
      c0 = wmma_bf16(a, b0, c0);
    }
    epi(c0, nt);
  }
}

// Persistent Bosh3 integrator: each workgroup owns M_TILE batch rows, keeps
// state + activations in LDS across all T steps.
__global__ void __launch_bounds__(NTHREADS)
node_bosh3_kernel(const float* __restrict__ x0, const float* __restrict__ ts,
                  const float* __restrict__ args,
                  const __bf16* __restrict__ Wp0, const __bf16* __restrict__ Wp1,
                  const __bf16* __restrict__ Wp2,
                  const float* __restrict__ bias0, const float* __restrict__ bias1,
                  const float* __restrict__ bias2,
                  float* __restrict__ out) {
  __shared__ float  yS[M_TILE][L_DIM];     // current state
  __shared__ float  accS[M_TILE][L_DIM];   // y_next accumulator
  __shared__ float  kS[M_TILE][L_DIM];     // current stage derivative
  __shared__ float  argS[M_TILE][P_DIM];
  __shared__ __bf16 actA[M_TILE][W_DIM];   // layer ping buffer (also stage input)
  __shared__ __bf16 actB[M_TILE][W_DIM];   // layer pong buffer

  const int tid  = threadIdx.x;
  const int wave = tid >> 5;
  const int lane = tid & 31;
  const int brow = blockIdx.x * M_TILE;

  // Init: async global->LDS copy of the x0 tile (gfx1250 async data mover),
  // then emit t=0 output from LDS; load args normally.
  {
    const float* src = x0 + (size_t)brow * L_DIM;
    uint32_t lds_base = (uint32_t)(uintptr_t)(&yS[0][0]);
    for (int i = tid; i < (M_TILE * L_DIM * 4) / 16; i += NTHREADS) {
      uint64_t ga = (uint64_t)(uintptr_t)src + (uint64_t)i * 16u;
      uint32_t la = lds_base + (uint32_t)i * 16u;
      asm volatile("global_load_async_to_lds_b128 %0, %1, off"
                   :: "v"(la), "v"(ga) : "memory");
    }
    asm volatile("s_wait_asynccnt 0x0" ::: "memory");
  }
  for (int e = tid; e < M_TILE * P_DIM; e += NTHREADS) {
    int r = e >> 3, c = e & (P_DIM - 1);
    argS[r][c] = args[(brow + r) * P_DIM + c];
  }
  __syncthreads();
  for (int e = tid; e < M_TILE * L_DIM; e += NTHREADS) {
    int r = e >> 7, c = e & (L_DIM - 1);
    out[(size_t)(brow + r) * T_DIM * L_DIM + c] = yS[r][c];
  }
  __syncthreads();

  for (int t = 1; t < T_DIM; ++t) {
    const float dt = ts[t] - ts[t - 1];

    // Build stage-1 input: actA = [y, args, 0-pad] as bf16
    for (int e = tid; e < M_TILE * K0PAD; e += NTHREADS) {
      int r = e / K0PAD, c = e % K0PAD;
      float v = (c < L_DIM) ? yS[r][c]
              : (c < L_DIM + P_DIM ? argS[r][c - L_DIM] : 0.f);
      actA[r][c] = (__bf16)v;
    }
    __syncthreads();

    for (int stage = 0; stage < 3; ++stage) {
      // 3-layer MLP: actA(K0PAD) -> actB(512) -> actA(512) -> kS(128, f32)
      mlp_layer<K0PAD, W_DIM, true, true>(&actA[0][0], W_DIM, Wp0, bias0,
                                          &actB[0][0], W_DIM, wave, lane);
      __syncthreads();
      mlp_layer<W_DIM, W_DIM, true, true>(&actB[0][0], W_DIM, Wp1, bias1,
                                          &actA[0][0], W_DIM, wave, lane);
      __syncthreads();
      mlp_layer<W_DIM, L_DIM, false, false>(&actA[0][0], W_DIM, Wp2, bias2,
                                            &kS[0][0], L_DIM, wave, lane);
      __syncthreads();

      if (stage == 0) {
        // k1: acc = y + dt*(2/9)k1 ; stage-2 input y + 0.5*dt*k1
        for (int e = tid; e < M_TILE * K0PAD; e += NTHREADS) {
          int r = e / K0PAD, c = e % K0PAD;
          if (c < L_DIM) {
            float y = yS[r][c], k = kS[r][c];
            accS[r][c] = y + dt * (2.f / 9.f) * k;
            actA[r][c] = (__bf16)(y + 0.5f * dt * k);
          } else {
            actA[r][c] = (c < L_DIM + P_DIM) ? (__bf16)argS[r][c - L_DIM]
                                             : (__bf16)0.f;
          }
        }
      } else if (stage == 1) {
        // k2: acc += dt*(1/3)k2 ; stage-3 input y + 0.75*dt*k2
        for (int e = tid; e < M_TILE * K0PAD; e += NTHREADS) {
          int r = e / K0PAD, c = e % K0PAD;
          if (c < L_DIM) {
            float y = yS[r][c], k = kS[r][c];
            accS[r][c] += dt * (1.f / 3.f) * k;
            actA[r][c] = (__bf16)(y + 0.75f * dt * k);
          } else {
            actA[r][c] = (c < L_DIM + P_DIM) ? (__bf16)argS[r][c - L_DIM]
                                             : (__bf16)0.f;
          }
        }
      } else {
        // k3: y_next = acc + dt*(4/9)k3 ; store output row t
        for (int e = tid; e < M_TILE * L_DIM; e += NTHREADS) {
          int r = e >> 7, c = e & (L_DIM - 1);
          float yn = accS[r][c] + dt * (4.f / 9.f) * kS[r][c];
          yS[r][c] = yn;
          out[(size_t)(brow + r) * T_DIM * L_DIM + (size_t)t * L_DIM + c] = yn;
        }
      }
      __syncthreads();
    }
  }
}

// One-time weight repack: f32 -> bf16, W0 K-padded 136 -> 160 with zeros.
__global__ void repack_kernel(const float* __restrict__ W0,
                              const float* __restrict__ W1,
                              const float* __restrict__ W2,
                              __bf16* __restrict__ wp) {
  const int n0 = W_DIM * K0PAD;
  const int n1 = W_DIM * W_DIM;
  const int n2 = L_DIM * W_DIM;
  int i = blockIdx.x * blockDim.x + threadIdx.x;
  if (i < n0) {
    int r = i / K0PAD, c = i % K0PAD;
    wp[i] = (c < (L_DIM + P_DIM)) ? (__bf16)W0[r * (L_DIM + P_DIM) + c]
                                  : (__bf16)0.f;
  } else if (i < n0 + n1) {
    int j = i - n0;
    wp[n0 + j] = (__bf16)W1[j];
  } else if (i < n0 + n1 + n2) {
    int j = i - n0 - n1;
    wp[n0 + n1 + j] = (__bf16)W2[j];
  }
}

extern "C" void kernel_launch(void* const* d_in, const int* in_sizes, int n_in,
                              void* d_out, int out_size, void* d_ws, size_t ws_size,
                              hipStream_t stream) {
  (void)in_sizes; (void)n_in; (void)out_size; (void)ws_size;
  const float* x0   = (const float*)d_in[0];
  const float* ts   = (const float*)d_in[1];
  const float* args = (const float*)d_in[2];
  const float* W0   = (const float*)d_in[3];
  const float* b0   = (const float*)d_in[4];
  const float* W1   = (const float*)d_in[5];
  const float* b1   = (const float*)d_in[6];
  const float* W2   = (const float*)d_in[7];
  const float* b2   = (const float*)d_in[8];
  float* out = (float*)d_out;

  __bf16* wp = (__bf16*)d_ws;
  const int n0 = W_DIM * K0PAD, n1 = W_DIM * W_DIM, n2 = L_DIM * W_DIM;
  const int total = n0 + n1 + n2;

  repack_kernel<<<(total + NTHREADS - 1) / NTHREADS, NTHREADS, 0, stream>>>(
      W0, W1, W2, wp);
  node_bosh3_kernel<<<B_TOT / M_TILE, NTHREADS, 0, stream>>>(
      x0, ts, args, wp, wp + n0, wp + n0 + n1, b0, b1, b2, out);
}
